// PointnetFPModule_enc_42125039239560
// MI455X (gfx1250) — matrix operand order repaired and checked by reading, not compiled, not run on hardware
//
#include <hip/hip_runtime.h>

typedef _Float16 v16h __attribute__((ext_vector_type(16)));
typedef float    v8f  __attribute__((ext_vector_type(8)));

#define B_   8
#define N_   8192
#define M_   2048
#define C_   256
#define K1_  512
#define D_   256
#define KE_  32

// ---------------- swizzled B-matrix layout helpers ----------------
// B fragment k order used by this code (contiguous, same as load_a):
// lane<16: element e -> B[k0+e][n0+lane]; lane>=16: e -> B[k0+16+e][n0+lane-16].
// We store F pre-swizzled so each lane's 16 halves are contiguous (32B/lane, 1KB/tile).
__device__ __forceinline__ size_t sw_idx(int b, int n, int k, int ktiles) {
  int nt = n >> 4, nr = n & 15, kt = k >> 5, kl = k & 31;
  return ((((size_t)b * (N_ >> 4) + nt) * ktiles + kt) << 9) +
         (size_t)((kl & 16) + nr) * 16 + (kl & 15);
}

// A fragment (16x32 f16, row-major source, lda in halves):
// lane<16 (m=lane): e -> K=e ; lane>=16: +16 on K.
__device__ __forceinline__ v16h load_a(const _Float16* __restrict__ A, int lda,
                                       int row, int k0, int lane) {
  int hsel = (lane >> 4) << 4;  // 0 or 16
  const _Float16* p = A + (size_t)row * lda + k0 + hsel;
  union { v16h v; uint4 u[2]; } r;
  r.u[0] = *reinterpret_cast<const uint4*>(p);
  r.u[1] = *reinterpret_cast<const uint4*>(p + 8);
  return r.v;
}

__device__ __forceinline__ v16h load_b(const _Float16* __restrict__ tile, int lane) {
  const _Float16* p = tile + lane * 16;
  union { v16h v; uint4 u[2]; } r;
  r.u[0] = *reinterpret_cast<const uint4*>(p);
  r.u[1] = *reinterpret_cast<const uint4*>(p + 8);
  return r.v;
}

// ---------------- K0: fold BN into conv weights ----------------
__global__ __launch_bounds__(256) void fold_kernel(
    const float* __restrict__ w1, const float* __restrict__ b1,
    const float* __restrict__ g1, const float* __restrict__ be1,
    const float* __restrict__ m1, const float* __restrict__ v1,
    const float* __restrict__ w2, const float* __restrict__ b2,
    const float* __restrict__ g2, const float* __restrict__ be2,
    const float* __restrict__ m2, const float* __restrict__ v2,
    _Float16* __restrict__ w1p, float* __restrict__ b1p,
    _Float16* __restrict__ w2p, float* __restrict__ b2p) {
  int t = blockIdx.x * 256 + threadIdx.x;
  {
    int o = t >> 9;
    float s = g1[o] * rsqrtf(v1[o] + 1e-5f);
    w1p[t] = (_Float16)(w1[t] * s);
    if ((t & 511) == 0) b1p[o] = s * (b1[o] - m1[o]) + be1[o];
  }
  if (t < D_ * D_) {
    int o = t >> 8;
    float s = g2[o] * rsqrtf(v2[o] + 1e-5f);
    w2p[t] = (_Float16)(w2[t] * s);
    if ((t & 255) == 0) b2p[o] = s * (b2[o] - m2[o]) + be2[o];
  }
}

// ---------------- K1a: three_nn (1 wave per unknown point) ----------------
__device__ __forceinline__ void ins3(float d, int i,
    float& d0, int& i0, float& d1, int& i1, float& d2, int& i2) {
  if (d < d2) {
    if (d < d1) {
      d2 = d1; i2 = i1;
      if (d < d0) { d1 = d0; i1 = i0; d0 = d; i0 = i; }
      else        { d1 = d;  i1 = i; }
    } else { d2 = d; i2 = i; }
  }
}

__global__ __launch_bounds__(256) void three_nn_kernel(
    const float* __restrict__ unknown, const float* __restrict__ known,
    int* __restrict__ idxb, float* __restrict__ wgtb) {
  int lane = threadIdx.x & 31;
  int gid  = blockIdx.x * 8 + (threadIdx.x >> 5);
  int b    = gid >> 13;
  const float* up = unknown + (size_t)gid * 3;
  float ux = up[0], uy = up[1], uz = up[2];
  float d0 = 1e30f, d1 = 1e30f, d2 = 1e30f;
  int   i0 = 0, i1 = 0, i2 = 0;
  const float* kb = known + (size_t)b * M_ * 3;
  for (int m = lane; m < M_; m += 32) {
    float dx = ux - kb[m * 3 + 0];
    float dy = uy - kb[m * 3 + 1];
    float dz = uz - kb[m * 3 + 2];
    float dd = fmaf(dx, dx, fmaf(dy, dy, dz * dz));
    ins3(dd, m, d0, i0, d1, i1, d2, i2);
  }
  for (int o = 16; o; o >>= 1) {
    float e0 = __shfl_xor(d0, o, 32), e1 = __shfl_xor(d1, o, 32), e2 = __shfl_xor(d2, o, 32);
    int   j0 = __shfl_xor(i0, o, 32), j1 = __shfl_xor(i1, o, 32), j2 = __shfl_xor(i2, o, 32);
    ins3(e0, j0, d0, i0, d1, i1, d2, i2);
    ins3(e1, j1, d0, i0, d1, i1, d2, i2);
    ins3(e2, j2, d0, i0, d1, i1, d2, i2);
  }
  if (!lane) {
    float s0 = sqrtf(fmaxf(d0, 1e-12f));
    float s1 = sqrtf(fmaxf(d1, 1e-12f));
    float s2 = sqrtf(fmaxf(d2, 1e-12f));
    float r0 = 1.f / (s0 + 1e-8f), r1 = 1.f / (s1 + 1e-8f), r2 = 1.f / (s2 + 1e-8f);
    float rs = r0 + r1 + r2;
    size_t p = (size_t)gid * 3;
    idxb[p + 0] = i0; idxb[p + 1] = i1; idxb[p + 2] = i2;
    wgtb[p + 0] = r0 / rs; wgtb[p + 1] = r1 / rs; wgtb[p + 2] = r2 / rs;
  }
}

// ---------------- K1b: interpolate + concat -> swizzled f16 feats ----------------
__global__ __launch_bounds__(256) void interp_concat_kernel(
    const float* __restrict__ kf, const float* __restrict__ uf,
    const int* __restrict__ idxb, const float* __restrict__ wgtb,
    _Float16* __restrict__ feats) {
  int b = blockIdx.y;
  int n = blockIdx.x * 256 + threadIdx.x;
  size_t p = (size_t)b * N_ + n;
  int   i0 = idxb[p * 3 + 0], i1 = idxb[p * 3 + 1], i2 = idxb[p * 3 + 2];
  float w0 = wgtb[p * 3 + 0], w1 = wgtb[p * 3 + 1], w2 = wgtb[p * 3 + 2];
  for (int c = 0; c < C_; ++c) {
    const float* row = kf + ((size_t)b * C_ + c) * M_;
    float v = fmaf(w0, row[i0], fmaf(w1, row[i1], w2 * row[i2]));
    feats[sw_idx(b, n, c, K1_ >> 5)] = (_Float16)v;
  }
  for (int c = 0; c < C_; ++c) {
    float v = uf[((size_t)b * C_ + c) * N_ + n];
    feats[sw_idx(b, n, C_ + c, K1_ >> 5)] = (_Float16)v;
  }
}

// ---------------- K2: GEMM1 (256x512 x 512x8192) + bias + ReLU -> swizzled f16 ----
// Block tile 128Mx128N (2 waves M x 4 waves N); wave tile 64Mx32N = 4x2 WMMA frags.
__global__ __launch_bounds__(256) void gemm1_kernel(
    const _Float16* __restrict__ W, const float* __restrict__ bias,
    const _Float16* __restrict__ F, _Float16* __restrict__ H1) {
  int lane = threadIdx.x & 31, wave = threadIdx.x >> 5;
  int wn = wave & 3, wm = wave >> 2;
  int b  = blockIdx.z;
  int m0 = blockIdx.y * 128 + wm * 64;
  int n0 = blockIdx.x * 128 + wn * 32;
  int mr = lane & 15, hf = lane >> 4;
  v8f acc[4][2] = {};
  for (int kt = 0; kt < (K1_ >> 5); ++kt) {
    int k0 = kt << 5;
    v16h a0 = load_a(W, K1_, m0 + mr, k0, lane);
    v16h a1 = load_a(W, K1_, m0 + 16 + mr, k0, lane);
    v16h a2 = load_a(W, K1_, m0 + 32 + mr, k0, lane);
    v16h a3 = load_a(W, K1_, m0 + 48 + mr, k0, lane);
    const _Float16* t0 = F + ((((size_t)b * (N_ >> 4) + (n0 >> 4)) * (K1_ >> 5) + kt) << 9);
    const _Float16* t1 = F + ((((size_t)b * (N_ >> 4) + (n0 >> 4) + 1) * (K1_ >> 5) + kt) << 9);
    v16h b0 = load_b(t0, lane);
    v16h b1 = load_b(t1, lane);
    acc[0][0] = __builtin_amdgcn_wmma_f32_16x16x32_f16(false, a0, false, b0, (short)0, acc[0][0], false, false);
    acc[0][1] = __builtin_amdgcn_wmma_f32_16x16x32_f16(false, a0, false, b1, (short)0, acc[0][1], false, false);
    acc[1][0] = __builtin_amdgcn_wmma_f32_16x16x32_f16(false, a1, false, b0, (short)0, acc[1][0], false, false);
    acc[1][1] = __builtin_amdgcn_wmma_f32_16x16x32_f16(false, a1, false, b1, (short)0, acc[1][1], false, false);
    acc[2][0] = __builtin_amdgcn_wmma_f32_16x16x32_f16(false, a2, false, b0, (short)0, acc[2][0], false, false);
    acc[2][1] = __builtin_amdgcn_wmma_f32_16x16x32_f16(false, a2, false, b1, (short)0, acc[2][1], false, false);
    acc[3][0] = __builtin_amdgcn_wmma_f32_16x16x32_f16(false, a3, false, b0, (short)0, acc[3][0], false, false);
    acc[3][1] = __builtin_amdgcn_wmma_f32_16x16x32_f16(false, a3, false, b1, (short)0, acc[3][1], false, false);
  }
  for (int mi = 0; mi < 4; ++mi)
    for (int ni = 0; ni < 2; ++ni)
      for (int r = 0; r < 8; ++r) {
        int m = m0 + mi * 16 + hf * 8 + r;
        int n = n0 + ni * 16 + mr;
        float v = acc[mi][ni][r] + bias[m];
        v = v > 0.f ? v : 0.f;
        H1[sw_idx(b, n, m, D_ >> 5)] = (_Float16)v;
      }
}

// ---------------- K3: GEMM2 (256x256 x 256x8192) + bias + ReLU -> f32 H2 ----------
__global__ __launch_bounds__(256) void gemm2_kernel(
    const _Float16* __restrict__ W, const float* __restrict__ bias,
    const _Float16* __restrict__ F, float* __restrict__ H2) {
  int lane = threadIdx.x & 31, wave = threadIdx.x >> 5;
  int wn = wave & 3, wm = wave >> 2;
  int b  = blockIdx.z;
  int m0 = blockIdx.y * 128 + wm * 64;
  int n0 = blockIdx.x * 128 + wn * 32;
  int mr = lane & 15, hf = lane >> 4;
  v8f acc[4][2] = {};
  for (int kt = 0; kt < (D_ >> 5); ++kt) {
    int k0 = kt << 5;
    v16h a0 = load_a(W, D_, m0 + mr, k0, lane);
    v16h a1 = load_a(W, D_, m0 + 16 + mr, k0, lane);
    v16h a2 = load_a(W, D_, m0 + 32 + mr, k0, lane);
    v16h a3 = load_a(W, D_, m0 + 48 + mr, k0, lane);
    const _Float16* t0 = F + ((((size_t)b * (N_ >> 4) + (n0 >> 4)) * (D_ >> 5) + kt) << 9);
    const _Float16* t1 = F + ((((size_t)b * (N_ >> 4) + (n0 >> 4) + 1) * (D_ >> 5) + kt) << 9);
    v16h b0 = load_b(t0, lane);
    v16h b1 = load_b(t1, lane);
    acc[0][0] = __builtin_amdgcn_wmma_f32_16x16x32_f16(false, a0, false, b0, (short)0, acc[0][0], false, false);
    acc[0][1] = __builtin_amdgcn_wmma_f32_16x16x32_f16(false, a0, false, b1, (short)0, acc[0][1], false, false);
    acc[1][0] = __builtin_amdgcn_wmma_f32_16x16x32_f16(false, a1, false, b0, (short)0, acc[1][0], false, false);
    acc[1][1] = __builtin_amdgcn_wmma_f32_16x16x32_f16(false, a1, false, b1, (short)0, acc[1][1], false, false);
    acc[2][0] = __builtin_amdgcn_wmma_f32_16x16x32_f16(false, a2, false, b0, (short)0, acc[2][0], false, false);
    acc[2][1] = __builtin_amdgcn_wmma_f32_16x16x32_f16(false, a2, false, b1, (short)0, acc[2][1], false, false);
    acc[3][0] = __builtin_amdgcn_wmma_f32_16x16x32_f16(false, a3, false, b0, (short)0, acc[3][0], false, false);
    acc[3][1] = __builtin_amdgcn_wmma_f32_16x16x32_f16(false, a3, false, b1, (short)0, acc[3][1], false, false);
  }
  for (int mi = 0; mi < 4; ++mi)
    for (int ni = 0; ni < 2; ++ni)
      for (int r = 0; r < 8; ++r) {
        int m = m0 + mi * 16 + hf * 8 + r;
        int n = n0 + ni * 16 + mr;
        float v = acc[mi][ni][r] + bias[m];
        v = v > 0.f ? v : 0.f;
        H2[((size_t)b * D_ + m) * N_ + n] = v;
      }
}

// ---------------- K4: soft-assignment softmax (1 wave/point, lane==codeword) ----------------
__global__ __launch_bounds__(256) void assign_kernel(
    const float* __restrict__ h2, const float* __restrict__ code,
    const float* __restrict__ scale, float* __restrict__ aout) {
  __shared__ float cs[KE_ * 257];
  __shared__ float c2s[KE_];
  for (int i = threadIdx.x; i < KE_ * D_; i += 256)
    cs[(i >> 8) * 257 + (i & 255)] = code[i];
  __syncthreads();
  if (threadIdx.x < KE_) {
    const float* r = &cs[threadIdx.x * 257];
    float s = 0.f;
    for (int d = 0; d < D_; ++d) s = fmaf(r[d], r[d], s);
    c2s[threadIdx.x] = s;
  }
  __syncthreads();
  int lane = threadIdx.x & 31, wave = threadIdx.x >> 5;
  int gid = blockIdx.x * 8 + wave;
  int b = gid >> 13, n = gid & (N_ - 1);
  const float* x  = h2 + (size_t)b * D_ * N_ + n;
  const float* cr = &cs[lane * 257];
  float dot = 0.f, xn = 0.f;
#pragma unroll 8
  for (int d = 0; d < D_; ++d) {
    float xv = x[(size_t)d * N_];        // same address across lanes -> broadcast
    dot = fmaf(xv, cr[d], dot);
    xn  = fmaf(xv, xv, xn);
  }
  float sl = scale[lane] * (xn - 2.f * dot + c2s[lane]);
  float mx = sl;
  for (int o = 16; o; o >>= 1) mx = fmaxf(mx, __shfl_xor(mx, o, 32));
  float ev = __expf(sl - mx);
  float sm = ev;
  for (int o = 16; o; o >>= 1) sm += __shfl_xor(sm, o, 32);
  aout[(size_t)gid * KE_ + lane] = ev / sm;   // (B,N,K) coalesced
}

// ---------------- K5: E = A^T X accumulation (atomics over n-segments) ----------------
__global__ __launch_bounds__(256) void aggregate_kernel(
    const float* __restrict__ aout, const float* __restrict__ h2,
    float* __restrict__ eacc, float* __restrict__ asum) {
  __shared__ float As[32 * 32];
  int b = blockIdx.x, seg = blockIdx.y;
  int d = threadIdx.x;
  const float* xrow = h2 + ((size_t)b * D_ + d) * N_;
  float acc[KE_];
#pragma unroll
  for (int k = 0; k < KE_; ++k) acc[k] = 0.f;
  float asr = 0.f;
  for (int c = 0; c < 16; ++c) {
    int n0 = seg * 512 + c * 32;
    __syncthreads();
    const float* ap = aout + ((size_t)b * N_ + n0) * KE_;
    for (int i = threadIdx.x; i < 1024; i += 256) As[i] = ap[i];
    __syncthreads();
    float xr[32];
    const float4* xv = (const float4*)(xrow + n0);
#pragma unroll
    for (int j = 0; j < 8; ++j) {
      float4 t = xv[j];
      xr[j * 4 + 0] = t.x; xr[j * 4 + 1] = t.y; xr[j * 4 + 2] = t.z; xr[j * 4 + 3] = t.w;
    }
#pragma unroll
    for (int j = 0; j < 32; ++j) {
      float xj = xr[j];
#pragma unroll
      for (int k = 0; k < KE_; ++k) acc[k] = fmaf(As[j * 32 + k], xj, acc[k]);
    }
    if (threadIdx.x < KE_) {
#pragma unroll
      for (int j = 0; j < 32; ++j) asr += As[j * 32 + threadIdx.x];
    }
  }
#pragma unroll
  for (int k = 0; k < KE_; ++k)
    atomicAdd(&eacc[((size_t)b * KE_ + k) * D_ + d], acc[k]);
  if (threadIdx.x < KE_) atomicAdd(&asum[b * KE_ + threadIdx.x], asr);
}

// ---------------- K6: E = relu(Eacc - Asum*code) ----------------
__global__ __launch_bounds__(256) void efinal_kernel(
    const float* __restrict__ eacc, const float* __restrict__ asum,
    const float* __restrict__ code, float* __restrict__ ebuf) {
  int i = blockIdx.x * 256 + threadIdx.x;          // < B*KE*D = 65536
  int b = i >> 13, r = i & 8191, k = r >> 8, d = r & 255;
  float v = eacc[i] - asum[b * KE_ + k] * code[k * D_ + d];
  ebuf[i] = v > 0.f ? v : 0.f;
}

// ---------------- K7: per-batch inverse L2 norm ----------------
__global__ __launch_bounds__(256) void norm_kernel(
    const float* __restrict__ ebuf, float* __restrict__ invn) {
  __shared__ float red[256];
  int b = blockIdx.x;
  float s = 0.f;
  for (int i = threadIdx.x; i < KE_ * D_; i += 256) {
    float v = ebuf[b * KE_ * D_ + i];
    s = fmaf(v, v, s);
  }
  red[threadIdx.x] = s; __syncthreads();
  for (int st = 128; st; st >>= 1) {
    if (threadIdx.x < st) red[threadIdx.x] += red[threadIdx.x + st];
    __syncthreads();
  }
  if (!threadIdx.x) invn[b] = 1.f / fmaxf(sqrtf(red[0]), 1e-12f);
}

// ---------------- K8: ctx = sigmoid((E/||E||) @ W^T + b) ----------------
__global__ __launch_bounds__(256) void ctx_kernel(
    const float* __restrict__ ebuf, const float* __restrict__ lw,
    const float* __restrict__ lb, const float* __restrict__ invn,
    float* __restrict__ ctx) {
  __shared__ float red[256];
  int d = blockIdx.x, b = blockIdx.y;
  const float* e = ebuf + (size_t)b * KE_ * D_;
  const float* w = lw + (size_t)d * KE_ * D_;
  float s = 0.f;
  for (int j = threadIdx.x; j < KE_ * D_; j += 256) s = fmaf(e[j], w[j], s);
  red[threadIdx.x] = s; __syncthreads();
  for (int st = 128; st; st >>= 1) {
    if (threadIdx.x < st) red[threadIdx.x] += red[threadIdx.x + st];
    __syncthreads();
  }
  if (!threadIdx.x) {
    float z = red[0] * invn[b] + lb[d];
    ctx[b * D_ + d] = 1.f / (1.f + __expf(-z));
  }
}

// ---------------- K9: out = H2 * ctx (channel attention) ----------------
__global__ __launch_bounds__(256) void scale_kernel(
    const float* __restrict__ h2, const float* __restrict__ ctx,
    float* __restrict__ out) {
  size_t i = (size_t)blockIdx.x * 256 + threadIdx.x;   // float4 index
  size_t e = i << 2;
  float c = ctx[e >> 13];                              // row = b*256 + d
  const float4* h4 = (const float4*)h2;
  float4 h = h4[i];
  float4* o4 = (float4*)out;
  o4[i] = make_float4(h.x * c, h.y * c, h.z * c, h.w * c);
}

extern "C" void kernel_launch(void* const* d_in, const int* in_sizes, int n_in,
                              void* d_out, int out_size, void* d_ws, size_t ws_size,
                              hipStream_t stream) {
  const float* unknown   = (const float*)d_in[0];
  const float* known     = (const float*)d_in[1];
  const float* uf        = (const float*)d_in[2];
  const float* kf        = (const float*)d_in[3];
  const float* w1        = (const float*)d_in[4];
  const float* b1        = (const float*)d_in[5];
  const float* g1        = (const float*)d_in[6];
  const float* be1       = (const float*)d_in[7];
  const float* m1        = (const float*)d_in[8];
  const float* v1        = (const float*)d_in[9];
  const float* w2        = (const float*)d_in[10];
  const float* b2        = (const float*)d_in[11];
  const float* g2        = (const float*)d_in[12];
  const float* be2       = (const float*)d_in[13];
  const float* m2        = (const float*)d_in[14];
  const float* v2        = (const float*)d_in[15];
  const float* code      = (const float*)d_in[16];
  const float* scale     = (const float*)d_in[17];
  const float* lw        = (const float*)d_in[18];
  const float* lb        = (const float*)d_in[19];
  float* out = (float*)d_out;

  char* ws = (char*)d_ws;
  _Float16* w1p   = (_Float16*)(ws + 0);           //  256*512*2 = 262144
  float*    b1p   = (float*)   (ws + 262144);      //  1024
  _Float16* w2p   = (_Float16*)(ws + 263168);      //  131072
  float*    b2p   = (float*)   (ws + 394240);      //  1024
  int*      idxb  = (int*)     (ws + 395264);      //  786432
  float*    wgtb  = (float*)   (ws + 1181696);     //  786432
  _Float16* feats = (_Float16*)(ws + 1968128);     //  67108864
  _Float16* h1    = (_Float16*)(ws + 69076992);    //  33554432
  float*    h2    = (float*)   (ws + 102631424);   //  67108864
  float*    aout  = (float*)   (ws + 169740288);   //  8388608
  float*    eacc  = (float*)   (ws + 178128896);   //  262144
  float*    asum  = (float*)   (ws + 178391040);   //  1024
  float*    ebuf  = (float*)   (ws + 178392064);   //  262144
  float*    invn  = (float*)   (ws + 178654208);   //  256
  float*    ctx   = (float*)   (ws + 178654464);   //  8192

  fold_kernel<<<512, 256, 0, stream>>>(w1, b1, g1, be1, m1, v1,
                                       w2, b2, g2, be2, m2, v2,
                                       w1p, b1p, w2p, b2p);
  three_nn_kernel<<<B_ * N_ / 8, 256, 0, stream>>>(unknown, known, idxb, wgtb);
  interp_concat_kernel<<<dim3(N_ / 256, B_), 256, 0, stream>>>(kf, uf, idxb, wgtb, feats);
  gemm1_kernel<<<dim3(N_ / 128, 2, B_), 256, 0, stream>>>(w1p, b1p, feats, h1);
  gemm2_kernel<<<dim3(N_ / 128, 2, B_), 256, 0, stream>>>(w2p, b2p, h1, h2);
  assign_kernel<<<B_ * N_ / 8, 256, 0, stream>>>(h2, code, scale, aout);
  hipMemsetAsync(eacc, 0, 262144 + 1024, stream);   // eacc + asum are contiguous
  aggregate_kernel<<<dim3(B_, 16), 256, 0, stream>>>(aout, h2, eacc, asum);
  efinal_kernel<<<B_ * KE_ * D_ / 256, 256, 0, stream>>>(eacc, asum, code, ebuf);
  norm_kernel<<<B_, 256, 0, stream>>>(ebuf, invn);
  ctx_kernel<<<dim3(D_, B_), 256, 0, stream>>>(ebuf, lw, lb, invn, ctx);
  scale_kernel<<<(B_ * D_ * N_ / 4) / 256, 256, 0, stream>>>(h2, ctx, out);
}
